// GRU_with_cells_category_86268713108351
// MI455X (gfx1250) — compile-verified, matrix operand order checked by reading, not run on hardware
//
#include <hip/hip_runtime.h>

typedef __attribute__((ext_vector_type(2))) float v2f;
typedef __attribute__((ext_vector_type(8))) float v8f;
typedef __attribute__((ext_vector_type(8))) __bf16 v8bf;
typedef __attribute__((ext_vector_type(16))) __bf16 v16bf;

#define B_ 32
#define T_ 128
#define H_ 1024
#define V_ 32000
#define BH (B_ * H_)

__device__ __forceinline__ v8f wmma_f32(v2f a, v2f b, v8f c) {
  // V_WMMA_F32_16X16X4_F32
  return __builtin_amdgcn_wmma_f32_16x16x4_f32(
      false, a, false, b, (short)0, c, false, false);
}

__device__ __forceinline__ v8f wmma_bf16(v16bf a, v16bf b, v8f c) {
  // V_WMMA_F32_16X16X32_BF16
  return __builtin_amdgcn_wmma_f32_16x16x32_bf16(
      false, a, false, b, (short)0, c, false, false);
}

// 16-bit A/B fragment: lane holds K = 8*lhalf..+7 (VGPR0-3) and
// K = 16+8*lhalf..+7 (VGPR4-7) of its row -> two contiguous b128 loads.
__device__ __forceinline__ v16bf load_frag16(const __bf16* rowp, int k, int lhalf) {
  union { v16bf f; struct { v8bf a, b; } h; } u;
  u.h.a = *(const v8bf*)(rowp + k + 8 * lhalf);
  u.h.b = *(const v8bf*)(rowp + k + 16 + 8 * lhalf);
  return u.f;
}

__device__ __forceinline__ float sigmoidf_(float x) {
  return 1.0f / (1.0f + __expf(-x));
}

// ---------------------------------------------------------------------------
__global__ __launch_bounds__(256) void embed_kernel(
    const int* __restrict__ tok, const int* __restrict__ cat,
    const float* __restrict__ wemb, const float* __restrict__ cemb,
    float* __restrict__ xseq) {
  int idx = blockIdx.x * blockDim.x + threadIdx.x;  // over B*T*H
  int h = idx % H_;
  int bt = idx / H_;
  int t = bt % T_;
  int b = bt / T_;
  int w = tok[b * T_ + t];
  int c = cat[b * T_ + t];
  float* row = xseq + ((size_t)t * B_ + b) * (2 * H_);
  row[h] = wemb[(size_t)w * H_ + h];
  row[H_ + h] = cemb[(size_t)c * H_ + h];
}

__global__ __launch_bounds__(256) void init_state_kernel(
    const float* __restrict__ h1, const float* __restrict__ h2,
    const float* __restrict__ h3, float* __restrict__ st) {
  int i = blockIdx.x * blockDim.x + threadIdx.x;  // over B*H
  st[0 * 2 * BH + i] = h1[i];
  st[1 * 2 * BH + i] = h2[i];
  st[2 * 2 * BH + i] = h3[i];
}

__global__ __launch_bounds__(256) void split_bf16_kernel(
    const float* __restrict__ in, __bf16* __restrict__ hi,
    __bf16* __restrict__ lo, const int n) {
  int i = blockIdx.x * blockDim.x + threadIdx.x;
  if (i < n) {
    float v = in[i];
    __bf16 h = (__bf16)v;
    hi[i] = h;
    lo[i] = (__bf16)(v - (float)h);
  }
}

// ---------------------------------------------------------------------------
// Fused GRU cell step, split-K across 8 waves, LDS reduction (f32 WMMA).
// ---------------------------------------------------------------------------
#define GRU_LOADS(S, KK)                                                     \
  a0##S = *(const v2f*)(A0 + (KK)); a1##S = *(const v2f*)(A1 + (KK));        \
  f0##S = *(const v2f*)(B0 + (KK)); f1##S = *(const v2f*)(B1 + (KK));        \
  f2##S = *(const v2f*)(B2 + (KK));
#define GRU_WMMA(S)                                                          \
  c00 = wmma_f32(a0##S, f0##S, c00); c01 = wmma_f32(a1##S, f0##S, c01);      \
  c10 = wmma_f32(a0##S, f1##S, c10); c11 = wmma_f32(a1##S, f1##S, c11);      \
  c20 = wmma_f32(a0##S, f2##S, c20); c21 = wmma_f32(a1##S, f2##S, c21);

__global__ __launch_bounds__(256) void gru_cell_kernel(
    const float* __restrict__ x, int ldx, int Kx,
    const float* __restrict__ gi_pre,            // [B, 3H] slice or null
    const float* __restrict__ h,
    const float* __restrict__ Wih, const float* __restrict__ Whh,
    const float* __restrict__ bih, const float* __restrict__ bhh,
    float* __restrict__ hout, float* __restrict__ cextra) {
  __shared__ float red[8][6][256];  // [wave][gate*2+mtile][16m x 16j]

  const int tid = threadIdx.x;
  const int w = __builtin_amdgcn_readfirstlane(tid >> 5);  // scalar wave id
  const int lane = tid & 31;
  const int lrow = lane & 15;
  const int lhalf = lane >> 4;
  const int j0 = blockIdx.x * 16;
  const bool has_pre = (gi_pre != nullptr);

  v8f c00 = {}, c01 = {}, c10 = {}, c11 = {}, c20 = {}, c21 = {};

  const float* A;
  const float* Wt;
  int ldA, ldW, kbeg, kend;
  if (has_pre) {
    A = h; ldA = H_; Wt = Whh; ldW = H_;
    kbeg = w * (H_ / 8); kend = kbeg + (H_ / 8);
  } else if (w < 4) {
    A = x; ldA = ldx; Wt = Wih; ldW = Kx;
    kbeg = w * (Kx / 4); kend = kbeg + (Kx / 4);
  } else {
    A = h; ldA = H_; Wt = Whh; ldW = H_;
    kbeg = (w - 4) * (H_ / 4); kend = kbeg + (H_ / 4);
  }
  kbeg = __builtin_amdgcn_readfirstlane(kbeg);
  kend = __builtin_amdgcn_readfirstlane(kend);

  const float* A0 = A + (size_t)lrow * ldA + 2 * lhalf;
  const float* A1 = A + (size_t)(16 + lrow) * ldA + 2 * lhalf;
  const float* B0 = Wt + (size_t)(0 * H_ + j0 + lrow) * ldW + 2 * lhalf;
  const float* B1 = Wt + (size_t)(1 * H_ + j0 + lrow) * ldW + 2 * lhalf;
  const float* B2 = Wt + (size_t)(2 * H_ + j0 + lrow) * ldW + 2 * lhalf;

  // ping-pong double buffering: no register rotation copies.
  // #k-steps = (kend-kbeg)/4 is even (>= 8) for every role.
  v2f a0p, a1p, f0p, f1p, f2p, a0q, a1q, f0q, f1q, f2q;
  GRU_LOADS(p, kbeg)
  int k = kbeg + 4;
  for (; k + 4 < kend; k += 8) {
    GRU_LOADS(q, k)
    GRU_WMMA(p)
    GRU_LOADS(p, k + 4)
    GRU_WMMA(q)
  }
  GRU_LOADS(q, k)   // k == kend-4 here
  GRU_WMMA(p)
  GRU_WMMA(q)

#pragma unroll
  for (int i = 0; i < 8; ++i) {
    const int off = (i + 8 * lhalf) * 16 + lrow;
    red[w][0][off] = c00[i]; red[w][1][off] = c01[i];
    red[w][2][off] = c10[i]; red[w][3][off] = c11[i];
    red[w][4][off] = c20[i]; red[w][5][off] = c21[i];
  }
  __syncthreads();

#pragma unroll
  for (int pass = 0; pass < 2; ++pass) {
    const int e = tid + pass * 256;
    const int jc = e & 15;
    const int m = e >> 4;  // batch row 0..31
    const int mt = m >> 4;
    const int off = (m & 15) * 16 + jc;
    const int j = j0 + jc;
    float gi0v, gi1v, gi2v, gh0v = 0.f, gh1v = 0.f, gh2v = 0.f;
    if (has_pre) {
#pragma unroll
      for (int ww = 0; ww < 8; ++ww) {
        gh0v += red[ww][0 + mt][off];
        gh1v += red[ww][2 + mt][off];
        gh2v += red[ww][4 + mt][off];
      }
      gi0v = gi_pre[(size_t)m * 3 * H_ + 0 * H_ + j];
      gi1v = gi_pre[(size_t)m * 3 * H_ + 1 * H_ + j];
      gi2v = gi_pre[(size_t)m * 3 * H_ + 2 * H_ + j];
    } else {
      gi0v = gi1v = gi2v = 0.f;
#pragma unroll
      for (int ww = 0; ww < 4; ++ww) {
        gi0v += red[ww][0 + mt][off];
        gi1v += red[ww][2 + mt][off];
        gi2v += red[ww][4 + mt][off];
      }
#pragma unroll
      for (int ww = 4; ww < 8; ++ww) {
        gh0v += red[ww][0 + mt][off];
        gh1v += red[ww][2 + mt][off];
        gh2v += red[ww][4 + mt][off];
      }
    }
    const float r = sigmoidf_(gi0v + gh0v + bih[j] + bhh[j]);
    const float z = sigmoidf_(gi1v + gh1v + bih[H_ + j] + bhh[H_ + j]);
    const float n = tanhf(gi2v + bih[2 * H_ + j] + r * (gh2v + bhh[2 * H_ + j]));
    const float hv = h[(size_t)m * H_ + j];
    const float o = (1.0f - z) * n + z * hv;
    hout[(size_t)m * H_ + j] = o;
    if (cextra) cextra[(size_t)m * H_ + j] = o;
  }
}

// ---------------------------------------------------------------------------
// f32 blocked GEMM (gi1 pre-GEMM + FC fallback), ping-pong pipelined.
// ---------------------------------------------------------------------------
#define GEMM_LOADS(S, KK)                                                    \
  a0##S = *(const v2f*)(A0 + (KK)); a1##S = *(const v2f*)(A1 + (KK));        \
  f0##S = *(const v2f*)(B0 + (KK)); f1##S = *(const v2f*)(B1 + (KK));        \
  f2##S = *(const v2f*)(B2 + (KK)); f3##S = *(const v2f*)(B3 + (KK));
#define GEMM_WMMA(S)                                                         \
  c00 = wmma_f32(a0##S, f0##S, c00); c01 = wmma_f32(a1##S, f0##S, c01);      \
  c10 = wmma_f32(a0##S, f1##S, c10); c11 = wmma_f32(a1##S, f1##S, c11);      \
  c20 = wmma_f32(a0##S, f2##S, c20); c21 = wmma_f32(a1##S, f2##S, c21);      \
  c30 = wmma_f32(a0##S, f3##S, c30); c31 = wmma_f32(a1##S, f3##S, c31);

__global__ __launch_bounds__(128) void gemm_kernel(
    const float* __restrict__ X, const int K,
    const float* __restrict__ W, const float* __restrict__ bias,
    float* __restrict__ out, const int N, const int remap) {
  const int wv = threadIdx.x >> 5;
  const int lane = threadIdx.x & 31;
  const int lrow = lane & 15;
  const int lhalf = lane >> 4;
  const int m0 = blockIdx.x * 32;
  const int n0 = blockIdx.y * 256 + wv * 64;

  v8f c00 = {}, c01 = {}, c10 = {}, c11 = {};
  v8f c20 = {}, c21 = {}, c30 = {}, c31 = {};

  const float* A0 = X + (size_t)(m0 + lrow) * K + 2 * lhalf;
  const float* A1 = X + (size_t)(m0 + 16 + lrow) * K + 2 * lhalf;
  const float* B0 = W + (size_t)(n0 + 0 * 16 + lrow) * K + 2 * lhalf;
  const float* B1 = W + (size_t)(n0 + 1 * 16 + lrow) * K + 2 * lhalf;
  const float* B2 = W + (size_t)(n0 + 2 * 16 + lrow) * K + 2 * lhalf;
  const float* B3 = W + (size_t)(n0 + 3 * 16 + lrow) * K + 2 * lhalf;

  v2f a0p, a1p, f0p, f1p, f2p, f3p, a0q, a1q, f0q, f1q, f2q, f3q;
  GEMM_LOADS(p, 0)
  int k = 4;
  for (; k + 4 < K; k += 8) {
    GEMM_LOADS(q, k)
    GEMM_WMMA(p)
    GEMM_LOADS(p, k + 4)
    GEMM_WMMA(q)
  }
  GEMM_LOADS(q, k)  // k == K-4
  GEMM_WMMA(p)
  GEMM_WMMA(q)

#define EPILOGUE(CT, TIDX, MT)                                               \
  {                                                                          \
    const int n = n0 + (TIDX)*16 + lrow;                                     \
    const float bb = bias ? bias[n] : 0.0f;                                  \
    _Pragma("unroll") for (int i = 0; i < 8; ++i) {                          \
      const int m = m0 + (MT)*16 + i + 8 * lhalf;                            \
      const float val = CT[i] + bb;                                          \
      if (remap) {                                                           \
        out[(size_t)(m & 31) * T_ * V_ + (size_t)(m >> 5) * V_ + n] = val;   \
      } else {                                                               \
        out[(size_t)m * N + n] = val;                                        \
      }                                                                      \
    }                                                                        \
  }
  EPILOGUE(c00, 0, 0) EPILOGUE(c01, 0, 1)
  EPILOGUE(c10, 1, 0) EPILOGUE(c11, 1, 1)
  EPILOGUE(c20, 2, 0) EPILOGUE(c21, 2, 1)
  EPILOGUE(c30, 3, 0) EPILOGUE(c31, 3, 1)
#undef EPILOGUE
}

// ---------------------------------------------------------------------------
// Split-precision bf16 FC GEMM with async-to-LDS A staging.
// A tile (32 rows x K, hi+lo) is shared by all 4 waves of the WG: it is
// copied into LDS with GLOBAL_LOAD_ASYNC_TO_LDS_B128 (ASYNCcnt) in two
// 64 KB half-K phases; fragment reads become ds_load_b128. B (fc_w) streams
// from global and stays L2-resident (131 MB < 192 MB L2).
// Grid (4096/32, 32000/256), block 128 (4 waves).
// ---------------------------------------------------------------------------
#define KH 512  // half-K per staging phase
__global__ __launch_bounds__(128) void fc_bf16_kernel(
    const __bf16* __restrict__ Ahi, const __bf16* __restrict__ Alo,
    const __bf16* __restrict__ Whi, const __bf16* __restrict__ Wlo,
    const float* __restrict__ bias, float* __restrict__ out) {
  __shared__ __bf16 ash[2 * 32 * KH];  // 64 KB: [hi|lo][row 0..31][KH]

  const int tid = threadIdx.x;
  const int wv = tid >> 5;
  const int lane = tid & 31;
  const int lrow = lane & 15;
  const int lhalf = lane >> 4;
  const int m0 = blockIdx.x * 32;
  const int n0 = blockIdx.y * 256 + wv * 64;

  v8f c00 = {}, c01 = {}, c10 = {}, c11 = {};
  v8f c20 = {}, c21 = {}, c30 = {}, c31 = {};

  const __bf16* b0h = Whi + (size_t)(n0 + 0 * 16 + lrow) * H_;
  const __bf16* b0l = Wlo + (size_t)(n0 + 0 * 16 + lrow) * H_;
  const __bf16* b1h = Whi + (size_t)(n0 + 1 * 16 + lrow) * H_;
  const __bf16* b1l = Wlo + (size_t)(n0 + 1 * 16 + lrow) * H_;
  const __bf16* b2h = Whi + (size_t)(n0 + 2 * 16 + lrow) * H_;
  const __bf16* b2l = Wlo + (size_t)(n0 + 2 * 16 + lrow) * H_;
  const __bf16* b3h = Whi + (size_t)(n0 + 3 * 16 + lrow) * H_;
  const __bf16* b3l = Wlo + (size_t)(n0 + 3 * 16 + lrow) * H_;

  const __bf16* a0h_l = ash + (size_t)(0 + lrow) * KH;         // hi rows
  const __bf16* a1h_l = ash + (size_t)(0 + 16 + lrow) * KH;
  const __bf16* a0l_l = ash + (size_t)(32 + lrow) * KH;        // lo rows
  const __bf16* a1l_l = ash + (size_t)(32 + 16 + lrow) * KH;
  const unsigned lds_base = (unsigned)(size_t)(&ash[0]);

  for (int p = 0; p < 2; ++p) {
    if (p) __syncthreads();  // all waves done reading previous phase
    // --- cooperative async fill: 4096 x b128 chunks, 32 per thread
    for (int i = 0; i < 32; ++i) {
      const int c = tid + 128 * i;          // chunk id 0..4095
      const int part = c >> 11;             // 0: hi, 1: lo
      const int rem = c & 2047;
      const int row = rem >> 6;             // KH*2/16 = 64 chunks per row
      const int colc = rem & 63;
      const __bf16* src = (part ? Alo : Ahi) +
                          (size_t)(m0 + row) * H_ + p * KH + colc * 8;
      const unsigned lds_addr = lds_base + (unsigned)c * 16;
      const unsigned long long ga = (unsigned long long)(size_t)src;
      asm volatile("global_load_async_to_lds_b128 %0, %1, off"
                   :: "v"(lds_addr), "v"(ga) : "memory");
    }
    asm volatile("s_wait_asynccnt 0" ::: "memory");
    __syncthreads();

    for (int k = 0; k < KH; k += 32) {
      const int kg = p * KH + k;  // global-K for B
      v16bf A0h = load_frag16(a0h_l, k, lhalf), A0l = load_frag16(a0l_l, k, lhalf);
      v16bf A1h = load_frag16(a1h_l, k, lhalf), A1l = load_frag16(a1l_l, k, lhalf);
      v16bf F0h = load_frag16(b0h, kg, lhalf), F0l = load_frag16(b0l, kg, lhalf);
      v16bf F1h = load_frag16(b1h, kg, lhalf), F1l = load_frag16(b1l, kg, lhalf);
      v16bf F2h = load_frag16(b2h, kg, lhalf), F2l = load_frag16(b2l, kg, lhalf);
      v16bf F3h = load_frag16(b3h, kg, lhalf), F3l = load_frag16(b3l, kg, lhalf);
      // hi*hi
      c00 = wmma_bf16(A0h, F0h, c00); c01 = wmma_bf16(A1h, F0h, c01);
      c10 = wmma_bf16(A0h, F1h, c10); c11 = wmma_bf16(A1h, F1h, c11);
      c20 = wmma_bf16(A0h, F2h, c20); c21 = wmma_bf16(A1h, F2h, c21);
      c30 = wmma_bf16(A0h, F3h, c30); c31 = wmma_bf16(A1h, F3h, c31);
      // hi*lo
      c00 = wmma_bf16(A0h, F0l, c00); c01 = wmma_bf16(A1h, F0l, c01);
      c10 = wmma_bf16(A0h, F1l, c10); c11 = wmma_bf16(A1h, F1l, c11);
      c20 = wmma_bf16(A0h, F2l, c20); c21 = wmma_bf16(A1h, F2l, c21);
      c30 = wmma_bf16(A0h, F3l, c30); c31 = wmma_bf16(A1h, F3l, c31);
      // lo*hi
      c00 = wmma_bf16(A0l, F0h, c00); c01 = wmma_bf16(A1l, F0h, c01);
      c10 = wmma_bf16(A0l, F1h, c10); c11 = wmma_bf16(A1l, F1h, c11);
      c20 = wmma_bf16(A0l, F2h, c20); c21 = wmma_bf16(A1l, F2h, c21);
      c30 = wmma_bf16(A0l, F3h, c30); c31 = wmma_bf16(A1l, F3h, c31);
    }
  }

#define EPILOGUE(CT, TIDX, MT)                                               \
  {                                                                          \
    const int n = n0 + (TIDX)*16 + lrow;                                     \
    const float bb = bias[n];                                                \
    _Pragma("unroll") for (int i = 0; i < 8; ++i) {                          \
      const int m = m0 + (MT)*16 + i + 8 * lhalf;                            \
      out[(size_t)(m & 31) * T_ * V_ + (size_t)(m >> 5) * V_ + n] =          \
          CT[i] + bb;                                                        \
    }                                                                        \
  }
  EPILOGUE(c00, 0, 0) EPILOGUE(c01, 0, 1)
  EPILOGUE(c10, 1, 0) EPILOGUE(c11, 1, 1)
  EPILOGUE(c20, 2, 0) EPILOGUE(c21, 2, 1)
  EPILOGUE(c30, 3, 0) EPILOGUE(c31, 3, 1)
#undef EPILOGUE
}

__global__ __launch_bounds__(256) void copy_hidden_kernel(
    const float* __restrict__ st, float* __restrict__ out) {
  int i = blockIdx.x * blockDim.x + threadIdx.x;  // over 3*B*H
  int layer = i / BH;
  int r = i % BH;
  out[i] = st[(size_t)layer * 2 * BH + r];  // final buffer index = T%2 = 0
}

extern "C" void kernel_launch(void* const* d_in, const int* in_sizes, int n_in,
                              void* d_out, int out_size, void* d_ws, size_t ws_size,
                              hipStream_t stream) {
  (void)in_sizes; (void)n_in; (void)out_size;
  const int* tok = (const int*)d_in[0];
  const int* cat = (const int*)d_in[1];
  const float* h1 = (const float*)d_in[3];
  const float* h2 = (const float*)d_in[4];
  const float* h3 = (const float*)d_in[5];
  const float* wemb = (const float*)d_in[6];
  const float* cemb = (const float*)d_in[7];
  const float* Wih[3] = {(const float*)d_in[8], (const float*)d_in[12], (const float*)d_in[16]};
  const float* Whh[3] = {(const float*)d_in[9], (const float*)d_in[13], (const float*)d_in[17]};
  const float* bih[3] = {(const float*)d_in[10], (const float*)d_in[14], (const float*)d_in[18]};
  const float* bhh[3] = {(const float*)d_in[11], (const float*)d_in[15], (const float*)d_in[19]};
  const float* fcw = (const float*)d_in[20];
  const float* fcb = (const float*)d_in[21];
  float* out = (float*)d_out;

  float* xseq = (float*)d_ws;
  float* c3seq = xseq + (size_t)T_ * B_ * 2 * H_;
  float* st = c3seq + (size_t)T_ * B_ * H_;
  float* gi1buf = st + 6 * (size_t)BH;
  const size_t f32_cnt = (size_t)T_ * B_ * 2 * H_ + (size_t)T_ * B_ * H_ +
                         6 * (size_t)BH + (size_t)T_ * B_ * 3 * H_;
  __bf16* c3hi = (__bf16*)(gi1buf + (size_t)T_ * B_ * 3 * H_);
  __bf16* c3lo = c3hi + (size_t)T_ * B_ * H_;
  __bf16* fchi = c3lo + (size_t)T_ * B_ * H_;
  __bf16* fclo = fchi + (size_t)V_ * H_;

  const size_t need_pre = f32_cnt * sizeof(float);
  const size_t need_bf16 =
      need_pre + (2 * (size_t)T_ * B_ * H_ + 2 * (size_t)V_ * H_) * sizeof(__bf16);
  const bool pre = (ws_size >= need_pre);
  const bool bf16fc = (ws_size >= need_bf16);

  embed_kernel<<<(B_ * T_ * H_) / 256, 256, 0, stream>>>(tok, cat, wemb, cemb, xseq);
  init_state_kernel<<<BH / 256, 256, 0, stream>>>(h1, h2, h3, st);

  if (pre) {
    dim3 g(4096 / 32, 3072 / 256);
    gemm_kernel<<<g, 128, 0, stream>>>(xseq, 2 * H_, Wih[0], nullptr, gi1buf,
                                       3 * H_, 0);
  }

  for (int t = 0; t < T_; ++t) {
    const int rd = t & 1, wr = rd ^ 1;
    float* s1r = st + 0 * 2 * BH + (size_t)rd * BH;
    float* s1w = st + 0 * 2 * BH + (size_t)wr * BH;
    float* s2r = st + 1 * 2 * BH + (size_t)rd * BH;
    float* s2w = st + 1 * 2 * BH + (size_t)wr * BH;
    float* s3r = st + 2 * 2 * BH + (size_t)rd * BH;
    float* s3w = st + 2 * 2 * BH + (size_t)wr * BH;
    const float* xt = xseq + (size_t)t * B_ * 2 * H_;
    const float* gp = pre ? (gi1buf + (size_t)t * B_ * 3 * H_) : nullptr;
    gru_cell_kernel<<<64, 256, 0, stream>>>(xt, 2 * H_, 2 * H_, gp, s1r,
                                            Wih[0], Whh[0], bih[0], bhh[0],
                                            s1w, nullptr);
    gru_cell_kernel<<<64, 256, 0, stream>>>(s1w, H_, H_, nullptr, s2r,
                                            Wih[1], Whh[1], bih[1], bhh[1],
                                            s2w, nullptr);
    gru_cell_kernel<<<64, 256, 0, stream>>>(s2w, H_, H_, nullptr, s3r,
                                            Wih[2], Whh[2], bih[2], bhh[2],
                                            s3w, c3seq + (size_t)t * B_ * H_);
  }

  copy_hidden_kernel<<<(3 * BH) / 256, 256, 0, stream>>>(st, out + (size_t)B_ * T_ * V_);

  if (bf16fc) {
    split_bf16_kernel<<<((size_t)T_ * B_ * H_) / 256, 256, 0, stream>>>(
        c3seq, c3hi, c3lo, T_ * B_ * H_);
    split_bf16_kernel<<<((size_t)V_ * H_) / 256, 256, 0, stream>>>(
        fcw, fchi, fclo, V_ * H_);
    dim3 fcg(4096 / 32, V_ / 256);  // (128, 125)
    fc_bf16_kernel<<<fcg, 128, 0, stream>>>(c3hi, c3lo, fchi, fclo, fcb, out);
  } else {
    dim3 fcg(4096 / 32, V_ / 256);
    gemm_kernel<<<fcg, 128, 0, stream>>>(c3seq, H_, fcw, fcb, out, V_, 1);
  }
}
